// GNN_noMMP_57174604644524
// MI455X (gfx1250) — matrix operand order repaired and checked by reading, not compiled, not run on hardware
//
#include <hip/hip_runtime.h>

#define HID 64
#define LAY 2

typedef _Float16 v16h __attribute__((ext_vector_type(16)));
typedef _Float16 v8h  __attribute__((ext_vector_type(8)));
typedef _Float16 h2t  __attribute__((ext_vector_type(2)));
typedef float v8f __attribute__((ext_vector_type(8)));
typedef float v4f __attribute__((ext_vector_type(4)));
typedef float v2f __attribute__((ext_vector_type(2)));

__device__ __forceinline__ v16h cat8(v8h lo, v8h hi) {
    return __builtin_shufflevector(lo, hi, 0,1,2,3,4,5,6,7,8,9,10,11,12,13,14,15);
}
__device__ __forceinline__ float silu(float x) { return x / (1.0f + __expf(-x)); }
__device__ __forceinline__ void atomic_add_f32(float* p, float v) {
    __hip_atomic_fetch_add(p, v, __ATOMIC_RELAXED, __HIP_MEMORY_SCOPE_AGENT);
}

// Transpose + f32->f16 convert a [K][N] weight into [N][K].
__global__ void convert_wt_kernel(const float* __restrict__ w, _Float16* __restrict__ wt,
                                  int K, int N) {
    int i = blockIdx.x * blockDim.x + threadIdx.x;
    if (i < K * N) {
        int k = i / N, n = i - k * N;
        wt[(size_t)n * K + k] = (_Float16)w[i];
    }
}

// ---------------- Edge MLP (+ fused scatter-add into agg) ----------------
// outT = W1T x inT (K=192) -> SiLU -> outT = W2T x hT (K=64)
__global__ __launch_bounds__(128) void edge_mlp_kernel(
    const float* __restrict__ ein, const float* __restrict__ xin,
    const int* __restrict__ src, const int* __restrict__ dst,
    const _Float16* __restrict__ w1t, const float* __restrict__ b1,
    const _Float16* __restrict__ w2t, const float* __restrict__ b2,
    float* __restrict__ eout, float* __restrict__ agg, int E)
{
    __shared__ __align__(16) _Float16 stage[4][16][192];  // 24 KB: cat(e, x[src], x[dst]) in f16
    __shared__ __align__(16) _Float16 hbuf[4][16][64];    //  8 KB: SiLU output in f16

    const int tid  = threadIdx.x;
    const int wave = tid >> 5;
    const int lane = tid & 31;
    const int half = lane >> 4;   // 0: lanes 0-15, 1: lanes 16-31
    const int l16  = lane & 15;

    const int tile  = blockIdx.x * 4 + wave;
    const bool valid = (tile + 1) * 16 <= E;
    const int eb = valid ? tile * 16 : 0;

    // ---- stage 16 rows of concatenated features (gather x[src], x[dst]) ----
    const int c0 = lane * 2;  // 0..62, covers 64 channels with float2 per lane
    for (int r = 0; r < 16; ++r) {
        const int e = eb + r;
        const int s = src[e];
        const int d = dst[e];
        v2f a  = *(const v2f*)(ein + (size_t)e * HID + c0);
        v2f bx = *(const v2f*)(xin + (size_t)s * HID + c0);
        v2f cx = *(const v2f*)(xin + (size_t)d * HID + c0);
        h2t p;
        p.x = (_Float16)a.x;  p.y = (_Float16)a.y;  *(h2t*)&stage[wave][r][c0]       = p;
        p.x = (_Float16)bx.x; p.y = (_Float16)bx.y; *(h2t*)&stage[wave][r][64 + c0]  = p;
        p.x = (_Float16)cx.x; p.y = (_Float16)cx.y; *(h2t*)&stage[wave][r][128 + c0] = p;
    }
    __syncthreads();

    // ---- GEMM1: acc[mt] = W1T(16x32 tile) x inT(32x16), K = 192 ----
    v8f acc[4] = {};
#pragma unroll
    for (int kb = 0; kb < 6; ++kb) {
        const _Float16* bp = &stage[wave][l16][kb * 32 + half * 8];
        v16h B = cat8(*(const v8h*)bp, *(const v8h*)(bp + 16));
#pragma unroll
        for (int mt = 0; mt < 4; ++mt) {
            const _Float16* ap = w1t + (size_t)(mt * 16 + l16) * 192 + kb * 32 + half * 8;
            v16h A = cat8(*(const v8h*)ap, *(const v8h*)(ap + 16));
            acc[mt] = __builtin_amdgcn_wmma_f32_16x16x32_f16(
                false, A, false, B, (short)0, acc[mt], false, false);
        }
    }

    // ---- bias + SiLU, repack h into f16 row-major [edge][channel] ----
#pragma unroll
    for (int mt = 0; mt < 4; ++mt) {
        const int m0 = mt * 16 + half * 8;
        v4f bA = *(const v4f*)(b1 + m0);
        v4f bB = *(const v4f*)(b1 + m0 + 4);
        float t[8];
#pragma unroll
        for (int j = 0; j < 4; ++j) {
            t[j]     = silu(acc[mt][j]     + bA[j]);
            t[j + 4] = silu(acc[mt][j + 4] + bB[j]);
        }
#pragma unroll
        for (int j = 0; j < 8; j += 2) {
            h2t p; p.x = (_Float16)t[j]; p.y = (_Float16)t[j + 1];
            *(h2t*)&hbuf[wave][l16][m0 + j] = p;
        }
    }
    __syncthreads();

    // ---- GEMM2: acc2[mt] = W2T(16x32 tile) x hT(32x16), K = 64 ----
    v8f acc2[4] = {};
#pragma unroll
    for (int kb = 0; kb < 2; ++kb) {
        const _Float16* bp = &hbuf[wave][l16][kb * 32 + half * 8];
        v16h B = cat8(*(const v8h*)bp, *(const v8h*)(bp + 16));
#pragma unroll
        for (int mt = 0; mt < 4; ++mt) {
            const _Float16* ap = w2t + (size_t)(mt * 16 + l16) * 64 + kb * 32 + half * 8;
            v16h A = cat8(*(const v8h*)ap, *(const v8h*)(ap + 16));
            acc2[mt] = __builtin_amdgcn_wmma_f32_16x16x32_f16(
                false, A, false, B, (short)0, acc2[mt], false, false);
        }
    }

    // ---- bias, store edge_attr_out, fused scatter-add into agg[dst] ----
    if (valid) {
        const int e = eb + l16;
        const int d = dst[e];
        float* orow = eout + (size_t)e * HID;
        float* arow = agg  + (size_t)d * HID;
#pragma unroll
        for (int mt = 0; mt < 4; ++mt) {
            const int m0 = mt * 16 + half * 8;
            v4f bA = *(const v4f*)(b2 + m0);
            v4f bB = *(const v4f*)(b2 + m0 + 4);
            v4f r0, r1;
#pragma unroll
            for (int j = 0; j < 4; ++j) {
                r0[j] = acc2[mt][j]     + bA[j];
                r1[j] = acc2[mt][j + 4] + bB[j];
            }
            *(v4f*)(orow + m0)     = r0;
            *(v4f*)(orow + m0 + 4) = r1;
#pragma unroll
            for (int j = 0; j < 4; ++j) {
                atomic_add_f32(arow + m0 + j,     r0[j]);
                atomic_add_f32(arow + m0 + 4 + j, r1[j]);
            }
        }
    }
}

// ---------------- Node MLP ----------------
// in = cat(x[n], agg[n]) [N,128]; outT = W1T x inT -> SiLU -> outT = W2T x hT
__global__ __launch_bounds__(128) void node_mlp_kernel(
    const float* __restrict__ xin, const float* __restrict__ agg,
    const _Float16* __restrict__ w1t, const float* __restrict__ b1,
    const _Float16* __restrict__ w2t, const float* __restrict__ b2,
    float* __restrict__ xout, int N)
{
    __shared__ __align__(16) _Float16 stage[4][16][128];  // 16 KB
    __shared__ __align__(16) _Float16 hbuf[4][16][64];    //  8 KB

    const int tid  = threadIdx.x;
    const int wave = tid >> 5;
    const int lane = tid & 31;
    const int half = lane >> 4;
    const int l16  = lane & 15;

    const int tile  = blockIdx.x * 4 + wave;
    const bool valid = (tile + 1) * 16 <= N;
    const int nb = valid ? tile * 16 : 0;

    const int c0 = lane * 2;
    for (int r = 0; r < 16; ++r) {
        const int n = nb + r;
        v2f a  = *(const v2f*)(xin + (size_t)n * HID + c0);
        v2f bx = *(const v2f*)(agg + (size_t)n * HID + c0);
        h2t p;
        p.x = (_Float16)a.x;  p.y = (_Float16)a.y;  *(h2t*)&stage[wave][r][c0]      = p;
        p.x = (_Float16)bx.x; p.y = (_Float16)bx.y; *(h2t*)&stage[wave][r][64 + c0] = p;
    }
    __syncthreads();

    v8f acc[4] = {};
#pragma unroll
    for (int kb = 0; kb < 4; ++kb) {
        const _Float16* bp = &stage[wave][l16][kb * 32 + half * 8];
        v16h B = cat8(*(const v8h*)bp, *(const v8h*)(bp + 16));
#pragma unroll
        for (int mt = 0; mt < 4; ++mt) {
            const _Float16* ap = w1t + (size_t)(mt * 16 + l16) * 128 + kb * 32 + half * 8;
            v16h A = cat8(*(const v8h*)ap, *(const v8h*)(ap + 16));
            acc[mt] = __builtin_amdgcn_wmma_f32_16x16x32_f16(
                false, A, false, B, (short)0, acc[mt], false, false);
        }
    }

#pragma unroll
    for (int mt = 0; mt < 4; ++mt) {
        const int m0 = mt * 16 + half * 8;
        v4f bA = *(const v4f*)(b1 + m0);
        v4f bB = *(const v4f*)(b1 + m0 + 4);
        float t[8];
#pragma unroll
        for (int j = 0; j < 4; ++j) {
            t[j]     = silu(acc[mt][j]     + bA[j]);
            t[j + 4] = silu(acc[mt][j + 4] + bB[j]);
        }
#pragma unroll
        for (int j = 0; j < 8; j += 2) {
            h2t p; p.x = (_Float16)t[j]; p.y = (_Float16)t[j + 1];
            *(h2t*)&hbuf[wave][l16][m0 + j] = p;
        }
    }
    __syncthreads();

    v8f acc2[4] = {};
#pragma unroll
    for (int kb = 0; kb < 2; ++kb) {
        const _Float16* bp = &hbuf[wave][l16][kb * 32 + half * 8];
        v16h B = cat8(*(const v8h*)bp, *(const v8h*)(bp + 16));
#pragma unroll
        for (int mt = 0; mt < 4; ++mt) {
            const _Float16* ap = w2t + (size_t)(mt * 16 + l16) * 64 + kb * 32 + half * 8;
            v16h A = cat8(*(const v8h*)ap, *(const v8h*)(ap + 16));
            acc2[mt] = __builtin_amdgcn_wmma_f32_16x16x32_f16(
                false, A, false, B, (short)0, acc2[mt], false, false);
        }
    }

    if (valid) {
        const int n = nb + l16;
        float* orow = xout + (size_t)n * HID;
#pragma unroll
        for (int mt = 0; mt < 4; ++mt) {
            const int m0 = mt * 16 + half * 8;
            v4f bA = *(const v4f*)(b2 + m0);
            v4f bB = *(const v4f*)(b2 + m0 + 4);
            v4f r0, r1;
#pragma unroll
            for (int j = 0; j < 4; ++j) {
                r0[j] = acc2[mt][j]     + bA[j];
                r1[j] = acc2[mt][j + 4] + bB[j];
            }
            *(v4f*)(orow + m0)     = r0;
            *(v4f*)(orow + m0 + 4) = r1;
        }
    }
}

extern "C" void kernel_launch(void* const* d_in, const int* in_sizes, int n_in,
                              void* d_out, int out_size, void* d_ws, size_t ws_size,
                              hipStream_t stream) {
    const float* x0   = (const float*)d_in[0];
    const float* e0   = (const float*)d_in[1];
    const int*   eidx = (const int*)d_in[2];
    const float* ew1  = (const float*)d_in[3];
    const float* eb1  = (const float*)d_in[4];
    const float* ew2  = (const float*)d_in[5];
    const float* eb2  = (const float*)d_in[6];
    const float* nw1  = (const float*)d_in[7];
    const float* nb1  = (const float*)d_in[8];
    const float* nw2  = (const float*)d_in[9];
    const float* nb2  = (const float*)d_in[10];

    const int N = in_sizes[0] / HID;   // B==1
    const int E = in_sizes[1] / HID;
    const int* src = eidx;
    const int* dst = eidx + E;

    // Output regions double as ping-pong buffers (in-place safe: each row's
    // output depends only on that row's input within a kernel).
    float* Xbuf = (float*)d_out;                       // [N][64] -> final x
    float* Ebuf = (float*)d_out + (size_t)N * HID;     // [E][64] -> final edge_attr

    // Workspace: agg (f32) + transposed f16 weights.
    float* agg = (float*)d_ws;
    const size_t aggBytes = (size_t)N * HID * sizeof(float);
    char* wsp = (char*)d_ws + ((aggBytes + 255) & ~(size_t)255);
    _Float16* w_et1 = (_Float16*)wsp;                       // [L][64][192]
    _Float16* w_et2 = w_et1 + (size_t)LAY * 64 * 192;       // [L][64][64]
    _Float16* w_nt1 = w_et2 + (size_t)LAY * 64 * 64;        // [L][64][128]
    _Float16* w_nt2 = w_nt1 + (size_t)LAY * 64 * 128;       // [L][64][64]

    for (int l = 0; l < LAY; ++l) {
        convert_wt_kernel<<<48, 256, 0, stream>>>(ew1 + (size_t)l * 192 * 64,
                                                  w_et1 + (size_t)l * 64 * 192, 192, 64);
        convert_wt_kernel<<<16, 256, 0, stream>>>(ew2 + (size_t)l * 64 * 64,
                                                  w_et2 + (size_t)l * 64 * 64, 64, 64);
        convert_wt_kernel<<<32, 256, 0, stream>>>(nw1 + (size_t)l * 128 * 64,
                                                  w_nt1 + (size_t)l * 64 * 128, 128, 64);
        convert_wt_kernel<<<16, 256, 0, stream>>>(nw2 + (size_t)l * 64 * 64,
                                                  w_nt2 + (size_t)l * 64 * 64, 64, 64);
    }

    const int eblocks = (E / 16 + 3) / 4;   // 4 waves/block, 16 edges/wave
    const int nblocks = (N / 16 + 3) / 4;

    for (int l = 0; l < LAY; ++l) {
        hipMemsetAsync(agg, 0, aggBytes, stream);
        const float* exin = (l == 0) ? x0 : Xbuf;
        const float* eein = (l == 0) ? e0 : Ebuf;
        edge_mlp_kernel<<<eblocks, 128, 0, stream>>>(
            eein, exin, src, dst,
            w_et1 + (size_t)l * 64 * 192, eb1 + l * HID,
            w_et2 + (size_t)l * 64 * 64,  eb2 + l * HID,
            Ebuf, agg, E);
        node_mlp_kernel<<<nblocks, 128, 0, stream>>>(
            exin, agg,
            w_nt1 + (size_t)l * 64 * 128, nb1 + l * HID,
            w_nt2 + (size_t)l * 64 * 64,  nb2 + l * HID,
            Xbuf, N);
    }
}
